// TorchLSTMCell_58703613002086
// MI455X (gfx1250) — matrix operand order repaired and checked by reading, not compiled, not run on hardware
//
#include <hip/hip_runtime.h>
#include <hip/hip_bf16.h>
#include <math.h>
#include <stdint.h>

// ---------------------------------------------------------------------------
// LSTM, persistent per-example-tile kernel for MI455X (gfx1250, wave32, WMMA)
//   gates[t] = [h | x_t] (16x384 bf16) @ Wcat^T (384x1024 bf16) + 2b
//   Wcat = [W_hh | W_ih] packed bf16, built once, stays hot in 192MB L2.
// 16 workgroups x 1024 threads; each WG owns 16 examples for all 512 steps.
// Weights stream from L2 each step (anti-LICM offset trick); K-tiles are
// double-buffered, with sched_group_barrier hints pinning the
// [vmem x4][ds x2][wmma x2] per-tile pipeline so waits stay partial.
// ---------------------------------------------------------------------------

typedef __attribute__((ext_vector_type(16))) __bf16 v16bf;
typedef __attribute__((ext_vector_type(8)))  __bf16 v8bf;
typedef __attribute__((ext_vector_type(4)))  __bf16 v4bf;
typedef __attribute__((ext_vector_type(8)))  float  v8f;
typedef __attribute__((ext_vector_type(4)))  float  f32x4;

#define N_EXD   256
#define N_IND   128
#define T_STEPS 512
#define HDIM    256
#define GDIM    1024          // 4*H
#define KCAT    384           // H + N_IN (concatenated GEMM K)
#define KTILES  (KCAT / 32)   // 12
#define EX_TILE 16
#define THREADS 1024
#define NBLOCKS (N_EXD / EX_TILE)  // 16

// sched_group_barrier masks
#define SGB_MFMA   0x008      // matrix ops (WMMA)
#define SGB_VMEMRD 0x020      // VMEM reads
#define SGB_DSRD   0x100      // DS reads

union AFrag { v16bf v; v8bf h[2]; };
union BFrag { v16bf v; v8bf h[2]; };

__device__ __forceinline__ float fast_sigmoid(float x) {
    return 1.0f / (1.0f + __expf(-x));
}
__device__ __forceinline__ float fast_tanh(float x) {
    // tanh(x) = 1 - 2/(exp(2x)+1); uses v_exp_f32 (trans VALU)
    float e = __expf(2.0f * x);
    return 1.0f - 2.0f / (e + 1.0f);
}

// ---- prep: pack Wcat[g][0:256] = W_hh[g][:], Wcat[g][256:384] = W_ih[g][:]
__global__ void build_wcat_kernel(const float* __restrict__ Whh,
                                  const float* __restrict__ Wih,
                                  __bf16* __restrict__ wcat) {
    int idx = blockIdx.x * blockDim.x + threadIdx.x;  // 0 .. 1024*384-1
    if (idx >= GDIM * KCAT) return;
    int g = idx / KCAT;
    int k = idx - g * KCAT;
    float v = (k < HDIM) ? Whh[g * HDIM + k] : Wih[g * N_IND + (k - HDIM)];
    wcat[idx] = (__bf16)v;
}

// ---- persistent LSTM kernel
__global__ __launch_bounds__(THREADS)
void lstm_persistent_kernel(const float*  __restrict__ X,      // [256,128,512]
                            const __bf16* __restrict__ wcat,   // [1024,384] bf16
                            const float*  __restrict__ bias,   // [1024]
                            float* __restrict__ outA,          // [512,256,256]
                            float* __restrict__ outC) {        // [512,256,256]
    __shared__ __bf16 ldsA[EX_TILE * KCAT];   // 12 KB   A = [h | x_t], row-major K
    __shared__ float  ldsG[EX_TILE * GDIM];   // 64 KB   gates f32
    __shared__ float  ldsB2[GDIM];            //  4 KB   2*b

    const int tid  = threadIdx.x;
    const int ex0  = blockIdx.x * EX_TILE;
    const int lane = tid & 31;
    const int wave = tid >> 5;        // 0..31
    const int m    = lane & 15;       // A-row / B-col within tile
    const int kh   = lane >> 4;       // K-half selector

    // init: A cleared (h starts at 0); bias*2 staged once
    for (int i = tid; i < EX_TILE * KCAT; i += THREADS) ldsA[i] = (__bf16)0.0f;
    ldsB2[tid] = 2.0f * bias[tid];

    // per-thread cell state: 4 consecutive (example, hidden) slots
    const int idx0 = tid * 4;
    const int bb   = idx0 >> 8;       // example within tile (0..15)
    const int jj   = idx0 & 255;      // hidden index (multiple of 4)
    float creg[4];
#pragma unroll
    for (int e = 0; e < 4; ++e) creg[e] = 0.0f;

    // Anti-LICM: opaque zero offset re-defined each iteration. The pointer
    // chain stays rooted at the kernel arg, so loads remain GLOBAL, but they
    // can no longer be hoisted out of the t loop into resident registers.
    size_t woff = 0;

    const __bf16* ldsArow = ldsA + m * KCAT;          // this lane's A row

    for (int t = 0; t < T_STEPS; ++t) {
        asm volatile("" : "+s"(woff));                // kill LICM on weight loads
        // wave owns N-tiles {2w, 2w+1}; this lane's two gate columns:
        const __bf16* wrow0 = wcat + woff + (size_t)(wave * 32 + m) * KCAT;
        const __bf16* wrow1 = wrow0 + (size_t)16 * KCAT;

        // ---- stage x_t into A columns [256,384): X[b][i][t], t-fastest layout
        for (int i = tid; i < EX_TILE * N_IND; i += THREADS) {
            int b  = i >> 7;
            int ii = i & 127;
            float v = X[(size_t)(ex0 + b) * (N_IND * T_STEPS) + (size_t)ii * T_STEPS + t];
            ldsA[b * KCAT + HDIM + ii] = (__bf16)v;
        }
        __syncthreads();   // h (prev step) + x_t ready; gates reads (prev step) done

        // ---- GEMM, K = 384 in 12 tiles of 32, double-buffered + sched hints
        v8f acc0 = {}, acc1 = {};
        AFrag a[2]; BFrag b0[2], b1[2];
        // prologue: K-tile 0
        a[0].h[0]  = *(const v8bf*)(ldsArow + kh * 16);
        a[0].h[1]  = *(const v8bf*)(ldsArow + kh * 16 + 8);
        b0[0].h[0] = *(const v8bf*)(wrow0 + kh * 16);
        b0[0].h[1] = *(const v8bf*)(wrow0 + kh * 16 + 8);
        b1[0].h[0] = *(const v8bf*)(wrow1 + kh * 16);
        b1[0].h[1] = *(const v8bf*)(wrow1 + kh * 16 + 8);
#pragma unroll
        for (int kt = 0; kt < KTILES; ++kt) {
            const int cur = kt & 1, nxt = cur ^ 1;
            if (kt < KTILES - 1) {     // issue loads for kt+1 before wmma(kt)
                const int ko = (kt + 1) * 32;
                b0[nxt].h[0] = *(const v8bf*)(wrow0 + ko + kh * 16);
                b0[nxt].h[1] = *(const v8bf*)(wrow0 + ko + kh * 16 + 8);
                b1[nxt].h[0] = *(const v8bf*)(wrow1 + ko + kh * 16);
                b1[nxt].h[1] = *(const v8bf*)(wrow1 + ko + kh * 16 + 8);
                a[nxt].h[0]  = *(const v8bf*)(ldsArow + ko + kh * 16);
                a[nxt].h[1]  = *(const v8bf*)(ldsArow + ko + kh * 16 + 8);
            }
            acc0 = __builtin_amdgcn_wmma_f32_16x16x32_bf16(
                     false, a[cur].v, false, b0[cur].v, (short)0, acc0, false, false);
            acc1 = __builtin_amdgcn_wmma_f32_16x16x32_bf16(
                     false, a[cur].v, false, b1[cur].v, (short)0, acc1, false, false);
            // pin per-tile pipeline: next-tile loads first, then this tile's wmmas
            __builtin_amdgcn_sched_group_barrier(SGB_VMEMRD, 4, 0);
            __builtin_amdgcn_sched_group_barrier(SGB_DSRD,   2, 0);
            __builtin_amdgcn_sched_group_barrier(SGB_MFMA,   2, 0);
        }
        // D scatter: VGPR r, lane l -> M=(l>>4)*8+r, N=l&15
        {
            const int ncol0 = wave * 32 + m;
#pragma unroll
            for (int r = 0; r < 8; ++r) {
                int mm = kh * 8 + r;
                ldsG[mm * GDIM + ncol0]      = acc0[r];
                ldsG[mm * GDIM + ncol0 + 16] = acc1[r];
            }
        }
        __syncthreads();   // gates complete

        // ---- elementwise: i,f,g,o -> c,h ; rewrite h (bf16) into A for t+1
        const float* gB = &ldsG[bb * GDIM];
        f32x4 g_i = *(const f32x4*)(gB + jj)            + *(const f32x4*)(ldsB2 + jj);
        f32x4 g_f = *(const f32x4*)(gB + HDIM   + jj)   + *(const f32x4*)(ldsB2 + HDIM   + jj);
        f32x4 g_g = *(const f32x4*)(gB + 2*HDIM + jj)   + *(const f32x4*)(ldsB2 + 2*HDIM + jj);
        f32x4 g_o = *(const f32x4*)(gB + 3*HDIM + jj)   + *(const f32x4*)(ldsB2 + 3*HDIM + jj);
        f32x4 hv, cv;
        v4bf  hb;
#pragma unroll
        for (int e = 0; e < 4; ++e) {
            float gi = fast_sigmoid(g_i[e]);
            float gf = fast_sigmoid(g_f[e]);
            float gg = fast_tanh   (g_g[e]);
            float go = fast_sigmoid(g_o[e]);
            float c  = gf * creg[e] + gi * gg;
            float h  = go * fast_tanh(c);
            creg[e]  = c;
            hv[e] = h;
            cv[e] = c;
            hb[e] = (__bf16)h;
        }
        *(v4bf*)(ldsA + bb * KCAT + jj) = hb;   // next step's A (h half), 8B store
        const size_t obase = (size_t)t * (N_EXD * HDIM)
                           + (size_t)(ex0 + bb) * HDIM + jj;
        *(f32x4*)(outA + obase) = hv;
        *(f32x4*)(outC + obase) = cv;
        // barrier at top of next iteration orders: h/x writes vs GEMM reads,
        // and gates reads (here) vs next GEMM's gates writes.
    }
}

extern "C" void kernel_launch(void* const* d_in, const int* in_sizes, int n_in,
                              void* d_out, int out_size, void* d_ws, size_t ws_size,
                              hipStream_t stream) {
    (void)in_sizes; (void)n_in; (void)out_size; (void)ws_size;
    const float* X   = (const float*)d_in[0];   // [256,128,512]
    const float* Wih = (const float*)d_in[1];   // [1024,128]
    const float* Whh = (const float*)d_in[2];   // [1024,256]
    const float* b   = (const float*)d_in[3];   // [1024]

    __bf16* wcat = (__bf16*)d_ws;               // 1024*384*2 = 768 KB scratch
    float*  outA = (float*)d_out;                                  // [512,256,256]
    float*  outC = outA + (size_t)T_STEPS * N_EXD * HDIM;          // [512,256,256]

    build_wcat_kernel<<<(GDIM * KCAT + 1023) / 1024, 1024, 0, stream>>>(Whh, Wih, wcat);
    lstm_persistent_kernel<<<NBLOCKS, THREADS, 0, stream>>>(X, wcat, b, outA, outC);
}
